// TransformerEncoder_15444702396491
// MI455X (gfx1250) — matrix operand order repaired and checked by reading, not compiled, not run on hardware
//
#include <hip/hip_runtime.h>

typedef __attribute__((ext_vector_type(16))) __bf16 bf16x16;
typedef __attribute__((ext_vector_type(8)))  __bf16 bf16x8;
typedef __attribute__((ext_vector_type(4)))  __bf16 bf16x4;
typedef __attribute__((ext_vector_type(8)))  float  floatx8;
typedef __attribute__((ext_vector_type(4)))  unsigned int u32x4;
typedef __attribute__((ext_vector_type(4)))  int i32x4;
typedef __attribute__((ext_vector_type(8)))  int i32x8;

union BFrag {
    bf16x16 v;
    bf16x8  h[2];
};

__device__ __forceinline__ floatx8 zero8() {
    floatx8 z;
#pragma unroll
    for (int i = 0; i < 8; ++i) z[i] = 0.0f;
    return z;
}

#if __has_builtin(__builtin_amdgcn_tensor_load_to_lds)
#define HAVE_TDM 1
#else
#define HAVE_TDM 0
#endif

#if HAVE_TDM
// Issue a TDM 2D tile load: rows x cols bf16, row-major, row stride in elems.
// D# layout per CDNA5 ISA ch.8: group0 = {count/type/addr}, group1 = dims.
__device__ __forceinline__ void tdm_load_tile_bf16(
    unsigned int lds_off, const __bf16* gptr, unsigned int rows,
    unsigned int cols, unsigned long long row_stride_elems) {
    unsigned long long ga = (unsigned long long)(uintptr_t)gptr;
    u32x4 g0;
    g0[0] = 1u;                                  // count=1, user mode
    g0[1] = lds_off;                             // lds_addr (bytes)
    g0[2] = (unsigned int)ga;                    // global_addr[95:64]
    g0[3] = (unsigned int)((ga >> 32) & 0x01FFFFFFu) | (2u << 30);  // +type=2
    i32x8 g1;
    g1[0] = (int)(1u << 16);                     // data_size=1 (2 bytes)
    g1[1] = (int)((cols & 0xFFFFu) << 16);       // tensor_dim0[15:0] @63:48
    g1[2] = (int)((cols >> 16) | ((rows & 0xFFFFu) << 16));  // d0 hi / d1 lo
    g1[3] = (int)((rows >> 16) | (cols << 16));  // d1 hi / tile_dim0
    g1[4] = (int)(rows & 0xFFFFu);               // tile_dim1, tile_dim2=0
    g1[5] = (int)(unsigned int)(row_stride_elems & 0xFFFFFFFFull);
    g1[6] = (int)(unsigned int)((row_stride_elems >> 32) & 0xFFFFull);
    g1[7] = 0;
    i32x4 z4 = {0, 0, 0, 0};
#if __clang_major__ >= 23
    i32x8 z8 = {0, 0, 0, 0, 0, 0, 0, 0};
    __builtin_amdgcn_tensor_load_to_lds(g0, g1, z4, z4, z8, 0);
#else
    __builtin_amdgcn_tensor_load_to_lds(g0, g1, z4, z4, 0);
#endif
}
#endif

// ---------------------------------------------------------------------------
// fp32 -> bf16 cast (weights), 4 elems/thread
// ---------------------------------------------------------------------------
__global__ __launch_bounds__(256) void cast_bf16_kernel(
    const float* __restrict__ in, __bf16* __restrict__ out, int n4) {
    int i = blockIdx.x * 256 + threadIdx.x;
    if (i < n4) {
        float4 f = ((const float4*)in)[i];
        bf16x4 o;
        o[0] = (__bf16)f.x; o[1] = (__bf16)f.y;
        o[2] = (__bf16)f.z; o[3] = (__bf16)f.w;
        ((bf16x4*)out)[i] = o;
    }
}

// ---------------------------------------------------------------------------
// LayerNorm over 1024 features, fp32 in -> bf16 out. One block per row.
// ---------------------------------------------------------------------------
__global__ __launch_bounds__(256) void ln_bf16_kernel(
    const float* __restrict__ x, const float* __restrict__ g,
    const float* __restrict__ b, __bf16* __restrict__ out) {
    __shared__ float red[256];
    const int row = blockIdx.x;
    const int tid = threadIdx.x;
    const float* xr = x + (size_t)row * 1024;

    float vals[4];
    float s = 0.0f;
#pragma unroll
    for (int i = 0; i < 4; ++i) { vals[i] = xr[tid + i * 256]; s += vals[i]; }
    red[tid] = s;
    __syncthreads();
    for (int off = 128; off > 0; off >>= 1) {
        if (tid < off) red[tid] += red[tid + off];
        __syncthreads();
    }
    const float mu = red[0] * (1.0f / 1024.0f);
    __syncthreads();

    float s2 = 0.0f;
#pragma unroll
    for (int i = 0; i < 4; ++i) { float d = vals[i] - mu; s2 += d * d; }
    red[tid] = s2;
    __syncthreads();
    for (int off = 128; off > 0; off >>= 1) {
        if (tid < off) red[tid] += red[tid + off];
        __syncthreads();
    }
    const float rstd = rsqrtf(red[0] * (1.0f / 1024.0f) + 1e-5f);

#pragma unroll
    for (int i = 0; i < 4; ++i) {
        int col = tid + i * 256;
        out[(size_t)row * 1024 + col] =
            (__bf16)((vals[i] - mu) * rstd * g[col] + b[col]);
    }
}

// ---------------------------------------------------------------------------
// C(MxN) = oscale*(A(MxK,bf16) @ B(NxK,bf16)^T + bias) [+relu] [+resid]
// Block 128x128, 8 waves (wave tile 32x64), K-tile 64, double-buffered LDS.
// Next tiles prefetched by the Tensor Data Mover while WMMAs run.
// ---------------------------------------------------------------------------
template <int RELU, int RESID, int OUTBF>
__global__ __launch_bounds__(256) void gemm_nt_bf16(
    const __bf16* __restrict__ A, const __bf16* __restrict__ B,
    const float* __restrict__ bias, const float* __restrict__ resid,
    float* __restrict__ outf, __bf16* __restrict__ outb,
    int M, int N, int K, float oscale) {
    __shared__ __bf16 As[2][128 * 64];
    __shared__ __bf16 Bs[2][128 * 64];

    const int tid   = threadIdx.x;
    const int lane  = tid & 31;
    const int wave  = tid >> 5;
    const int lhalf = lane >> 4;
    const int l16   = lane & 15;
    const int bm = blockIdx.y * 128;
    const int bn = blockIdx.x * 128;
    const int wm = (wave >> 1) * 32;
    const int wn = (wave & 1) * 64;

    floatx8 acc[2][4];
#pragma unroll
    for (int i = 0; i < 2; ++i)
#pragma unroll
        for (int j = 0; j < 4; ++j) acc[i][j] = zero8();

#if HAVE_TDM
    if (wave == 0) {
        tdm_load_tile_bf16((unsigned int)(uintptr_t)&As[0][0],
                           &A[(size_t)bm * K], 128, 64, (unsigned long long)K);
        tdm_load_tile_bf16((unsigned int)(uintptr_t)&Bs[0][0],
                           &B[(size_t)bn * K], 128, 64, (unsigned long long)K);
        __builtin_amdgcn_s_wait_tensorcnt(0);
    }
    __syncthreads();
#endif

    int buf = 0;
    for (int k0 = 0; k0 < K; k0 += 64) {
#if HAVE_TDM
        const int knext = k0 + 64;
        if (knext < K && wave == 0) {
            tdm_load_tile_bf16((unsigned int)(uintptr_t)&As[buf ^ 1][0],
                               &A[(size_t)bm * K + knext], 128, 64,
                               (unsigned long long)K);
            tdm_load_tile_bf16((unsigned int)(uintptr_t)&Bs[buf ^ 1][0],
                               &B[(size_t)bn * K + knext], 128, 64,
                               (unsigned long long)K);
        }
#else
        // Manual staging fallback: 128x64 bf16 per matrix, 16B chunks.
#pragma unroll
        for (int c = tid; c < 1024; c += 256) {
            int r  = c >> 3;
            int cc = (c & 7) * 8;
            *(bf16x8*)&As[buf][r * 64 + cc] =
                *(const bf16x8*)&A[(size_t)(bm + r) * K + k0 + cc];
            *(bf16x8*)&Bs[buf][r * 64 + cc] =
                *(const bf16x8*)&B[(size_t)(bn + r) * K + k0 + cc];
        }
        __syncthreads();
#endif
#pragma unroll
        for (int ks = 0; ks < 2; ++ks) {
            BFrag af[2];
#pragma unroll
            for (int i = 0; i < 2; ++i) {
                int mrow = wm + i * 16 + l16;
                af[i].h[0] =
                    *(const bf16x8*)&As[buf][mrow * 64 + ks * 32 + lhalf * 8];
                af[i].h[1] =
                    *(const bf16x8*)&As[buf][mrow * 64 + ks * 32 + lhalf * 8 + 16];
            }
            BFrag bfr[4];
#pragma unroll
            for (int j = 0; j < 4; ++j) {
                int nrow = wn + j * 16 + l16;
                bfr[j].h[0] =
                    *(const bf16x8*)&Bs[buf][nrow * 64 + ks * 32 + lhalf * 8];
                bfr[j].h[1] =
                    *(const bf16x8*)&Bs[buf][nrow * 64 + ks * 32 + lhalf * 8 + 16];
            }
#pragma unroll
            for (int i = 0; i < 2; ++i)
#pragma unroll
                for (int j = 0; j < 4; ++j)
                    acc[i][j] = __builtin_amdgcn_wmma_f32_16x16x32_bf16(
                        false, af[i].v, false, bfr[j].v, (short)0, acc[i][j],
                        false, false);
        }
#if HAVE_TDM
        if (wave == 0) __builtin_amdgcn_s_wait_tensorcnt(0);
#endif
        __syncthreads();
        buf ^= 1;
    }

    // Epilogue: C layout -> VGPR v holds (M = base + lhalf*8 + v, N = l16)
#pragma unroll
    for (int i = 0; i < 2; ++i) {
#pragma unroll
        for (int j = 0; j < 4; ++j) {
            const int col = bn + wn + j * 16 + l16;
            const float bia = bias[col];
#pragma unroll
            for (int v = 0; v < 8; ++v) {
                const int row = bm + wm + i * 16 + lhalf * 8 + v;
                float val = (acc[i][j][v] + bia) * oscale;
                if (RELU) val = fmaxf(val, 0.0f);
                if (RESID) val += resid[(size_t)row * N + col];
                if (OUTBF)
                    outb[(size_t)row * N + col] = (__bf16)val;
                else
                    outf[(size_t)row * N + col] = val;
            }
        }
    }
}

// ---------------------------------------------------------------------------
// Flash attention: one block per (128 q-rows, batch*head). 8 waves x 16 rows.
// Q is pre-scaled by 1/sqrt(1024) in the Q-projection epilogue.
// K tile arrives via TDM; V tile is transposed into LDS by the waves.
// ---------------------------------------------------------------------------
__global__ __launch_bounds__(256) void attn_kernel(
    const __bf16* __restrict__ q, const __bf16* __restrict__ k,
    const __bf16* __restrict__ v, __bf16* __restrict__ out) {
    __shared__ __bf16 Ks[64 * 64];      // [key][d]
    __shared__ __bf16 Vt[64 * 64];      // [d][key]
    __shared__ __bf16 Ps[8][16 * 64];   // per-wave P tile [row][key]

    const int tid   = threadIdx.x;
    const int lane  = tid & 31;
    const int wave  = tid >> 5;
    const int lhalf = lane >> 4;
    const int l16   = lane & 15;

    const int qblk = blockIdx.x;
    const int bh   = blockIdx.y;
    const int bat  = bh >> 4;
    const int h    = bh & 15;
    const size_t rowbase = (size_t)bat * 2048;
    const int hoff  = h * 64;
    const int qrow0 = qblk * 128 + wave * 16;

    BFrag qf[2];
    {
        const __bf16* qp = q + (rowbase + qrow0 + l16) * 1024 + hoff;
#pragma unroll
        for (int s = 0; s < 2; ++s) {
            qf[s].h[0] = *(const bf16x8*)&qp[s * 32 + lhalf * 8];
            qf[s].h[1] = *(const bf16x8*)&qp[s * 32 + lhalf * 8 + 16];
        }
    }

    floatx8 oacc[4];
#pragma unroll
    for (int dt = 0; dt < 4; ++dt) oacc[dt] = zero8();
    float mrun[8], lrun[8];
#pragma unroll
    for (int vv = 0; vv < 8; ++vv) { mrun[vv] = -3.0e38f; lrun[vv] = 0.0f; }

    for (int j0 = 0; j0 < 2048; j0 += 64) {
        __syncthreads();
#if HAVE_TDM
        if (wave == 0)
            tdm_load_tile_bf16((unsigned int)(uintptr_t)&Ks[0],
                               &k[(rowbase + j0) * 1024 + hoff], 64, 64, 1024ull);
#else
#pragma unroll
        for (int c = tid; c < 512; c += 256) {
            int r  = c >> 3;
            int cc = (c & 7) * 8;
            *(bf16x8*)&Ks[r * 64 + cc] =
                *(const bf16x8*)&k[(rowbase + j0 + r) * 1024 + hoff + cc];
        }
#endif
        // Stage transposed V tile: Vt[d][key]
        {
            int jj = tid & 63;
            int d0 = (tid >> 6) * 16;
            const __bf16* vp = v + (rowbase + j0 + jj) * 1024 + hoff + d0;
#pragma unroll
            for (int e = 0; e < 16; ++e) Vt[(d0 + e) * 64 + jj] = vp[e];
        }
#if HAVE_TDM
        if (wave == 0) __builtin_amdgcn_s_wait_tensorcnt(0);
#endif
        __syncthreads();

        // S = Q * K^T (Q pre-scaled): 4 key-tiles of 16, 2 k-steps over d
        floatx8 sacc[4];
#pragma unroll
        for (int nt = 0; nt < 4; ++nt) {
            sacc[nt] = zero8();
#pragma unroll
            for (int ks = 0; ks < 2; ++ks) {
                BFrag bfr;
                int nrow = nt * 16 + l16;
                bfr.h[0] = *(const bf16x8*)&Ks[nrow * 64 + ks * 32 + lhalf * 8];
                bfr.h[1] = *(const bf16x8*)&Ks[nrow * 64 + ks * 32 + lhalf * 8 + 16];
                sacc[nt] = __builtin_amdgcn_wmma_f32_16x16x32_bf16(
                    false, qf[ks].v, false, bfr.v, (short)0, sacc[nt],
                    false, false);
            }
        }

        // Online softmax per row (row = lhalf*8 + vv; 16 lanes per row group)
        float mnew[8], corr[8];
#pragma unroll
        for (int vv = 0; vv < 8; ++vv) {
            float tm = -3.0e38f;
#pragma unroll
            for (int nt = 0; nt < 4; ++nt) tm = fmaxf(tm, sacc[nt][vv]);
#pragma unroll
            for (int msk = 8; msk >= 1; msk >>= 1)
                tm = fmaxf(tm, __shfl_xor(tm, msk, 32));
            mnew[vv] = fmaxf(mrun[vv], tm);
            corr[vv] = __expf(mrun[vv] - mnew[vv]);
            mrun[vv] = mnew[vv];
        }
#pragma unroll
        for (int vv = 0; vv < 8; ++vv) {
            float ps = 0.0f;
#pragma unroll
            for (int nt = 0; nt < 4; ++nt) {
                float e = __expf(sacc[nt][vv] - mnew[vv]);
                sacc[nt][vv] = e;
                ps += e;
            }
#pragma unroll
            for (int msk = 8; msk >= 1; msk >>= 1)
                ps += __shfl_xor(ps, msk, 32);
            lrun[vv] = lrun[vv] * corr[vv] + ps;
#pragma unroll
            for (int dt = 0; dt < 4; ++dt) oacc[dt][vv] *= corr[vv];
        }

        // P tile -> wave-private LDS (bf16): C-layout -> A-layout relayout
        __bf16* Pw = &Ps[wave][0];
#pragma unroll
        for (int nt = 0; nt < 4; ++nt)
#pragma unroll
            for (int vv = 0; vv < 8; ++vv)
                Pw[(lhalf * 8 + vv) * 64 + nt * 16 + l16] = (__bf16)sacc[nt][vv];
        asm volatile("s_wait_dscnt 0" ::: "memory");

        // O += P * V : 2 k-steps over keys, 4 d-tiles
#pragma unroll
        for (int ks = 0; ks < 2; ++ks) {
            BFrag pf;
            pf.h[0] = *(const bf16x8*)&Pw[l16 * 64 + ks * 32 + lhalf * 8];
            pf.h[1] = *(const bf16x8*)&Pw[l16 * 64 + ks * 32 + lhalf * 8 + 16];
#pragma unroll
            for (int dt = 0; dt < 4; ++dt) {
                BFrag vf;
                int dcol = dt * 16 + l16;
                vf.h[0] = *(const bf16x8*)&Vt[dcol * 64 + ks * 32 + lhalf * 8];
                vf.h[1] = *(const bf16x8*)&Vt[dcol * 64 + ks * 32 + lhalf * 8 + 16];
                oacc[dt] = __builtin_amdgcn_wmma_f32_16x16x32_bf16(
                    false, pf.v, false, vf.v, (short)0, oacc[dt], false, false);
            }
        }
    }

#pragma unroll
    for (int dt = 0; dt < 4; ++dt) {
#pragma unroll
        for (int vv = 0; vv < 8; ++vv) {
            int row = qrow0 + lhalf * 8 + vv;
            int col = hoff + dt * 16 + l16;
            out[(rowbase + row) * 1024 + col] = (__bf16)(oacc[dt][vv] / lrun[vv]);
        }
    }
}

// ---------------------------------------------------------------------------
extern "C" void kernel_launch(void* const* d_in, const int* in_sizes, int n_in,
                              void* d_out, int out_size, void* d_ws,
                              size_t ws_size, hipStream_t stream) {
    const float* x     = (const float*)d_in[0];
    const float* ln1_g = (const float*)d_in[1];
    const float* ln1_b = (const float*)d_in[2];
    const float* wq    = (const float*)d_in[3];
    const float* bq    = (const float*)d_in[4];
    const float* wk    = (const float*)d_in[5];
    const float* bk    = (const float*)d_in[6];
    const float* wv    = (const float*)d_in[7];
    const float* bv    = (const float*)d_in[8];
    const float* wo    = (const float*)d_in[9];
    const float* bo    = (const float*)d_in[10];
    const float* ln2_g = (const float*)d_in[11];
    const float* ln2_b = (const float*)d_in[12];
    const float* w1    = (const float*)d_in[13];
    const float* b1    = (const float*)d_in[14];
    const float* w2    = (const float*)d_in[15];
    const float* b2    = (const float*)d_in[16];

    const int M = 4096;  // 2 * 2048 tokens
    char* ws = (char*)d_ws;
    size_t off = 0;
    auto alloc = [&](size_t bytes) {
        void* p = ws + off;
        off += (bytes + 255) & ~(size_t)255;
        return p;
    };
    __bf16* hbf   = (__bf16*)alloc((size_t)M * 1024 * 2);
    __bf16* wqb   = (__bf16*)alloc((size_t)1024 * 1024 * 2);
    __bf16* wkb   = (__bf16*)alloc((size_t)1024 * 1024 * 2);
    __bf16* wvb   = (__bf16*)alloc((size_t)1024 * 1024 * 2);
    __bf16* wob   = (__bf16*)alloc((size_t)1024 * 1024 * 2);
    __bf16* w1b   = (__bf16*)alloc((size_t)4096 * 1024 * 2);
    __bf16* w2b   = (__bf16*)alloc((size_t)1024 * 4096 * 2);
    __bf16* qb    = (__bf16*)alloc((size_t)M * 1024 * 2);
    __bf16* kb    = (__bf16*)alloc((size_t)M * 1024 * 2);
    __bf16* vb    = (__bf16*)alloc((size_t)M * 1024 * 2);
    __bf16* attnb = (__bf16*)alloc((size_t)M * 1024 * 2);
    float*  out1  = (float*)alloc((size_t)M * 1024 * 4);
    __bf16* h2b   = (__bf16*)alloc((size_t)M * 1024 * 2);
    __bf16* m1b   = (__bf16*)alloc((size_t)M * 4096 * 2);

    // 1) Cast weights to bf16
    {
        int n4 = 1024 * 1024 / 4;
        cast_bf16_kernel<<<(n4 + 255) / 256, 256, 0, stream>>>(wq, wqb, n4);
        cast_bf16_kernel<<<(n4 + 255) / 256, 256, 0, stream>>>(wk, wkb, n4);
        cast_bf16_kernel<<<(n4 + 255) / 256, 256, 0, stream>>>(wv, wvb, n4);
        cast_bf16_kernel<<<(n4 + 255) / 256, 256, 0, stream>>>(wo, wob, n4);
        int n4m = 4096 * 1024 / 4;
        cast_bf16_kernel<<<(n4m + 255) / 256, 256, 0, stream>>>(w1, w1b, n4m);
        cast_bf16_kernel<<<(n4m + 255) / 256, 256, 0, stream>>>(w2, w2b, n4m);
    }

    // 2) LN1 -> bf16
    ln_bf16_kernel<<<M, 256, 0, stream>>>(x, ln1_g, ln1_b, hbf);

    // 3) Q,K,V projections (bf16 out). Q pre-scaled by 1/sqrt(1024).
    {
        dim3 g(1024 / 128, M / 128);
        gemm_nt_bf16<0, 0, 1><<<g, 256, 0, stream>>>(
            hbf, wqb, bq, nullptr, nullptr, qb, M, 1024, 1024, 0.03125f);
        gemm_nt_bf16<0, 0, 1><<<g, 256, 0, stream>>>(
            hbf, wkb, bk, nullptr, nullptr, kb, M, 1024, 1024, 1.0f);
        gemm_nt_bf16<0, 0, 1><<<g, 256, 0, stream>>>(
            hbf, wvb, bv, nullptr, nullptr, vb, M, 1024, 1024, 1.0f);
    }

    // 4) Attention
    {
        dim3 ga(2048 / 128, 32);
        attn_kernel<<<ga, 256, 0, stream>>>(qb, kb, vb, attnb);
    }

    // 5) Output projection + residual x -> out1 (fp32)
    {
        dim3 g(1024 / 128, M / 128);
        gemm_nt_bf16<0, 1, 0><<<g, 256, 0, stream>>>(
            attnb, wob, bo, x, out1, nullptr, M, 1024, 1024, 1.0f);
    }

    // 6) LN2 -> bf16
    ln_bf16_kernel<<<M, 256, 0, stream>>>(out1, ln2_g, ln2_b, h2b);

    // 7) MLP hidden: relu(h2 @ w1^T + b1) -> bf16
    {
        dim3 g(4096 / 128, M / 128);
        gemm_nt_bf16<1, 0, 1><<<g, 256, 0, stream>>>(
            h2b, w1b, b1, nullptr, nullptr, m1b, M, 4096, 1024, 1.0f);
    }

    // 8) MLP out: relu(m1 @ w2^T + b2) + out1 -> d_out (fp32)
    {
        dim3 g(1024 / 128, M / 128);
        gemm_nt_bf16<1, 1, 0><<<g, 256, 0, stream>>>(
            m1b, w2b, b2, out1, (float*)d_out, nullptr, M, 1024, 4096, 1.0f);
    }
}
